// HypergraphAttentionLayer_19215683682839
// MI455X (gfx1250) — matrix operand order repaired and checked by reading, not compiled, not run on hardware
//
#include <hip/hip_runtime.h>
#include <hip/hip_bf16.h>
#include <stdint.h>

// ---------- problem constants ----------
#define BDIM 2
#define NND  8192   // nodes (N)
#define NE   4096   // edges (E)
#define FIN  256
#define NH   8
#define DOUT 64
#define HD   512    // NH*DOUT
#define ALPHA 0.2f

// ---------- types (own ext-vectors: union/POD safe) ----------
typedef __bf16 bf16_t;
typedef __bf16 v16bf __attribute__((ext_vector_type(16)));
typedef float  v8f   __attribute__((ext_vector_type(8)));
typedef float  f32x4 __attribute__((ext_vector_type(4)));
typedef unsigned int u32x4 __attribute__((ext_vector_type(4)));
typedef unsigned int u32x2 __attribute__((ext_vector_type(2)));
typedef int i32x4 __attribute__((ext_vector_type(4)));
typedef int i32x8 __attribute__((ext_vector_type(8)));

union FragU { v16bf v; u32x4 q[2]; };
union Pack8 { bf16_t h[8]; u32x4 q; };
union Pack4 { bf16_t h[4]; u32x2 u; };

#define LDT 40  // LDS tile row stride in halves (32 + 8 pad; 80B keeps 16B align)
#define TILE_HALVES (128 * LDT)

#if defined(__has_builtin)
#if __has_builtin(__builtin_amdgcn_tensor_load_to_lds)
#define HAVE_TDM 1
#endif
#endif
#ifndef HAVE_TDM
#define HAVE_TDM 0
#endif

#if HAVE_TDM
// 2D tensor_load_to_lds: tile_d1 rows x tile_d0 elements (bf16), row stride
// stride0 elements in global memory. TDM pad: after every 16 DWORDs (64B =
// one 32-half row) insert 4 DWORDs (16B) -> LDS rows at 80B = LDT halves,
// matching the WMMA fragment loaders exactly.
__device__ __forceinline__ void tdm_load_2d(unsigned lds_byte_addr, const void* gaddr,
                                            unsigned tensor_d0, unsigned tensor_d1,
                                            unsigned long long stride0,
                                            unsigned tile_d0, unsigned tile_d1) {
  unsigned long long ga = (unsigned long long)(uintptr_t)gaddr;
  u32x4 g0;
  g0.x = 0x1u;                                              // count=1, user mode
  g0.y = lds_byte_addr;                                     // lds_addr
  g0.z = (unsigned)(ga & 0xFFFFFFFFu);                      // global_addr[31:0]
  g0.w = (unsigned)((ga >> 32) & 0x1FFFFFFu) | (2u << 30);  // ga[56:32] | type=2
  i32x8 g1;
  g1[0] = (int)((1u << 16) |        // data_size = 2B
                (1u << 20) |        // pad_enable
                (3u << 22) |        // pad_interval: 16 DWORDs
                (3u << 25));        // pad_amount: 4 DWORDs
  g1[1] = (int)((tensor_d0 & 0xFFFFu) << 16);
  g1[2] = (int)(((tensor_d0 >> 16) & 0xFFFFu) | ((tensor_d1 & 0xFFFFu) << 16));
  g1[3] = (int)(((tensor_d1 >> 16) & 0xFFFFu) | ((tile_d0 & 0xFFFFu) << 16));
  g1[4] = (int)(tile_d1 & 0xFFFFu);                         // tile_dim1 | tile_dim2=0
  g1[5] = (int)(unsigned)(stride0 & 0xFFFFFFFFu);
  g1[6] = (int)(unsigned)((stride0 >> 32) & 0xFFFFu);       // stride0[47:32] | stride1=0
  g1[7] = 0;
  i32x4 gz = {0, 0, 0, 0};
#if __clang_major__ >= 23
  i32x8 gz8 = {0, 0, 0, 0, 0, 0, 0, 0};
  __builtin_amdgcn_tensor_load_to_lds(g0, g1, gz, gz, gz8, 0);
#else
  __builtin_amdgcn_tensor_load_to_lds(g0, g1, gz, gz, 0);
#endif
}
#endif  // HAVE_TDM

// A fragment (16x32 bf16): lane<16: row=lane, K 0..7 in q0, K 16..23 in q1.
// lane>=16: row=lane-16, K 8..15 in q0, K 24..31 in q1.  LDS layout As[m][k].
__device__ __forceinline__ v16bf load_frag_a(const bf16_t* As, int m0, int lane) {
  int l16 = lane & 15, hi = lane >> 4;
  const bf16_t* p = As + (m0 + l16) * LDT + hi * 8;
  FragU f;
  f.q[0] = *(const u32x4*)(p);
  f.q[1] = *(const u32x4*)(p + 16);
  return f.v;
}
// B fragment (32x16 bf16): lane<16: col=lane, K 0..7 in q0, K 16..23 in q1; lane>=16: col=lane-16, K 8..15 in q0, K 24..31 in q1.
__device__ __forceinline__ v16bf load_frag_b(const bf16_t* Bs, int n0, int lane) {
  int l16 = lane & 15, hi = lane >> 4;
  const bf16_t* p = Bs + (n0 + l16) * LDT + hi * 8;
  FragU f;
  f.q[0] = *(const u32x4*)(p);
  f.q[1] = *(const u32x4*)(p + 16);
  return f.v;
}

// ---------- K0a: H f32 -> bf16 ----------
__global__ void k_cvt_H(const float* __restrict__ H, bf16_t* __restrict__ Hb) {
  size_t i = (size_t)blockIdx.x * 1024 + (size_t)threadIdx.x * 4;
  f32x4 v = *(const f32x4*)(H + i);
  Pack4 t;
  t.h[0] = (bf16_t)v.x; t.h[1] = (bf16_t)v.y; t.h[2] = (bf16_t)v.z; t.h[3] = (bf16_t)v.w;
  *(u32x2*)(Hb + i) = t.u;
}

// ---------- K0b: W f32 (FIN x HD) -> WTb bf16 (HD x FIN) ----------
__global__ void k_cvt_Wt(const float* __restrict__ W, bf16_t* __restrict__ WTb) {
  int t = blockIdx.x * 256 + threadIdx.x;   // FIN*HD = 131072
  int f = t >> 9, hd = t & 511;
  WTb[hd * FIN + f] = (bf16_t)W[t];
}

// ---------- K1: Whb[b] = x[b] (f32->bf16) @ W ; also store WhbT ----------
__global__ void k_gemm_xW(const float* __restrict__ x, const bf16_t* __restrict__ WTb,
                          bf16_t* __restrict__ Whb, bf16_t* __restrict__ WhbT) {
  __shared__ __align__(16) bf16_t As[TILE_HALVES];
  __shared__ __align__(16) bf16_t Bs[TILE_HALVES];
  const int b = blockIdx.z;
  const int m0blk = blockIdx.y * 128;   // node rows
  const int n0blk = blockIdx.x * 128;   // hd cols
  const int tid = threadIdx.x, lane = tid & 31, wv = tid >> 5;
  const int wm = wv >> 2, wn = wv & 3;
  const v8f vz = {0.f,0.f,0.f,0.f,0.f,0.f,0.f,0.f};
  v8f acc[4][2];
#pragma unroll
  for (int i = 0; i < 4; ++i)
#pragma unroll
    for (int j = 0; j < 2; ++j) acc[i][j] = vz;
  const float* xb = x + (size_t)b * NND * FIN;

  for (int kt = 0; kt < FIN / 32; ++kt) {
    int k0 = kt * 32;
#pragma unroll
    for (int r = 0; r < 4; ++r) {           // A: 128x32 f32 -> bf16, contiguous k
      int L = tid + r * 256, row = L >> 3, fg = L & 7;
      f32x4 v = *(const f32x4*)(xb + (size_t)(m0blk + row) * FIN + k0 + fg * 4);
      Pack4 t;
      t.h[0] = (bf16_t)v.x; t.h[1] = (bf16_t)v.y; t.h[2] = (bf16_t)v.z; t.h[3] = (bf16_t)v.w;
      *(u32x2*)(&As[row * LDT + fg * 4]) = t.u;
    }
#pragma unroll
    for (int r = 0; r < 2; ++r) {           // B: WTb rows hd, contiguous k
      int L = tid + r * 256, row = L >> 2, kg = L & 3;
      u32x4 q = *(const u32x4*)(WTb + (size_t)(n0blk + row) * FIN + k0 + kg * 8);
      *(u32x4*)(&Bs[row * LDT + kg * 8]) = q;
    }
    __syncthreads();
    v16bf af[4], bfr[2];
#pragma unroll
    for (int i = 0; i < 4; ++i) af[i] = load_frag_a(As, wm * 64 + i * 16, lane);
#pragma unroll
    for (int j = 0; j < 2; ++j) bfr[j] = load_frag_b(Bs, wn * 32 + j * 16, lane);
#pragma unroll
    for (int i = 0; i < 4; ++i)
#pragma unroll
      for (int j = 0; j < 2; ++j)
        acc[i][j] = __builtin_amdgcn_wmma_f32_16x16x32_bf16(
            false, af[i], false, bfr[j], (short)0, acc[i][j], false, false);
    __syncthreads();
  }
  int l16 = lane & 15, hi = lane >> 4;
#pragma unroll
  for (int i = 0; i < 4; ++i)
#pragma unroll
    for (int j = 0; j < 2; ++j) {
      int col = n0blk + wn * 32 + j * 16 + l16;
      int m_base = m0blk + wm * 64 + i * 16 + hi * 8;
      Pack8 t;
#pragma unroll
      for (int r = 0; r < 8; ++r) {
        t.h[r] = (bf16_t)acc[i][j][r];
        Whb[((size_t)b * NND + m_base + r) * HD + col] = t.h[r];
      }
      *(u32x4*)(WhbT + ((size_t)b * HD + col) * NND + m_base) = t.q;
    }
}

// ---------- K1b: av = leaky_relu(Wh . a) -> avb[n][b*8+h] bf16 ----------
__global__ void k_av(const bf16_t* __restrict__ Whb, const float* __restrict__ a,
                     bf16_t* __restrict__ avb) {
  __shared__ float s_a[DOUT];
  if (threadIdx.x < DOUT) s_a[threadIdx.x] = a[threadIdx.x];
  __syncthreads();
  int t = blockIdx.x * 256 + threadIdx.x;   // BDIM*NND*NH = 131072
  int h = t & 7, n = (t >> 3) & (NND - 1), b = t >> 16;
  const bf16_t* p = Whb + ((size_t)b * NND + n) * HD + h * DOUT;
  float s = 0.f;
#pragma unroll
  for (int g = 0; g < 8; ++g) {
    u32x4 q = *(const u32x4*)(p + g * 8);
    const bf16_t* hh = (const bf16_t*)&q;
#pragma unroll
    for (int j = 0; j < 8; ++j) s += (float)hh[j] * s_a[g * 8 + j];
  }
  float v = s > 0.f ? s : ALPHA * s;
  avb[n * 16 + b * 8 + h] = (bf16_t)v;
}

// ---------- K2: logitsT[bh][e] = (H^T @ avb), WMMA M=E N=16 K=NND ----------
__global__ void k_gemm_logits(const bf16_t* __restrict__ Hb, const bf16_t* __restrict__ avb,
                              float* __restrict__ logitsT) {
  __shared__ __align__(16) bf16_t As[TILE_HALVES];
  __shared__ __align__(16) bf16_t Bs[16 * LDT];
  const int e0 = blockIdx.y * 128;
  const int tid = threadIdx.x, lane = tid & 31, wv = tid >> 5;
  const v8f vz = {0.f,0.f,0.f,0.f,0.f,0.f,0.f,0.f};
  v8f acc = vz;

  for (int kt = 0; kt < NND / 32; ++kt) {
    int k0 = kt * 32;
#pragma unroll
    for (int r = 0; r < 2; ++r) {           // A = H^T: coalesced read, b16 scatter
      int L = tid + r * 256, n_l = L >> 4, eg = L & 15;
      u32x4 q = *(const u32x4*)(Hb + (size_t)(k0 + n_l) * NE + e0 + eg * 8);
      const bf16_t* hh = (const bf16_t*)&q;
#pragma unroll
      for (int j = 0; j < 8; ++j) As[(eg * 8 + j) * LDT + n_l] = hh[j];
    }
#pragma unroll
    for (int r = 0; r < 2; ++r) {           // B: 32k x 16col from avb[n][16]
      int L = tid + r * 256, kl = L >> 4, c = L & 15;
      if (kl < 32) Bs[c * LDT + kl] = avb[(size_t)(k0 + kl) * 16 + c];
    }
    __syncthreads();
    v16bf af = load_frag_a(As, wv * 16, lane);
    v16bf bfr = load_frag_b(Bs, 0, lane);
    acc = __builtin_amdgcn_wmma_f32_16x16x32_bf16(false, af, false, bfr, (short)0, acc,
                                                  false, false);
    __syncthreads();
  }
  int l16 = lane & 15, hi = lane >> 4;
  int col = l16;                             // bh = b*8+h
  int e_base = e0 + wv * 16 + hi * 8;
#pragma unroll
  for (int r = 0; r < 8; ++r) logitsT[(size_t)col * NE + e_base + r] = acc[r];
}

// ---------- K3: softmax over E per (b,h); ae[b][e][h] ----------
__global__ void k_softmax(const float* __restrict__ logitsT, float* __restrict__ ae) {
  const int bh = blockIdx.x, b = bh >> 3, h = bh & 7;
  const float* row = logitsT + (size_t)bh * NE;
  __shared__ float red[256];
  float mx = -1e30f;
  for (int e = threadIdx.x; e < NE; e += 256) mx = fmaxf(mx, row[e]);
  red[threadIdx.x] = mx; __syncthreads();
  for (int s = 128; s > 0; s >>= 1) {
    if (threadIdx.x < s) red[threadIdx.x] = fmaxf(red[threadIdx.x], red[threadIdx.x + s]);
    __syncthreads();
  }
  mx = red[0]; __syncthreads();
  float sum = 0.f;
  for (int e = threadIdx.x; e < NE; e += 256) sum += __expf(row[e] - mx);
  red[threadIdx.x] = sum; __syncthreads();
  for (int s = 128; s > 0; s >>= 1) {
    if (threadIdx.x < s) red[threadIdx.x] += red[threadIdx.x + s];
    __syncthreads();
  }
  float inv = 1.f / red[0];
  for (int e = threadIdx.x; e < NE; e += 256)
    ae[((size_t)b * NE + e) * NH + h] = __expf(row[e] - mx) * inv;
}

// ---------- K4: mscT[b][hd][e] = bf16( ae[b][e][h] * (H^T @ Whb)[e][hd] ) ----------
__global__ void k_gemm_m(const bf16_t* __restrict__ Hb, const bf16_t* __restrict__ WhbT,
                         const float* __restrict__ ae, bf16_t* __restrict__ mscT) {
  __shared__ __align__(16) bf16_t As[TILE_HALVES];
  __shared__ __align__(16) bf16_t Bs[TILE_HALVES];
  const int b = blockIdx.z;
  const int e0blk = blockIdx.y * 128;   // M = edges
  const int n0blk = blockIdx.x * 128;   // N = hd
  const int tid = threadIdx.x, lane = tid & 31, wv = tid >> 5;
  const int wm = wv >> 2, wn = wv & 3;
  const v8f vz = {0.f,0.f,0.f,0.f,0.f,0.f,0.f,0.f};
  v8f acc[4][2];
#pragma unroll
  for (int i = 0; i < 4; ++i)
#pragma unroll
    for (int j = 0; j < 2; ++j) acc[i][j] = vz;
  const int KT = NND / 32;
  const bf16_t* Bsrc = WhbT + ((size_t)b * HD + n0blk) * NND;

  for (int kt = 0; kt < KT; ++kt) {
    int k0 = kt * 32;
#if HAVE_TDM
    if (wv == 0)    // B tile via Tensor Data Mover, overlapped with A scatter fill
      tdm_load_2d((unsigned)(uintptr_t)Bs, Bsrc + k0, NND, HD, NND, 32, 128);
#endif
#pragma unroll
    for (int r = 0; r < 2; ++r) {           // A = H^T tile: coalesced along e, b16 scatter
      int L = tid + r * 256, n_l = L >> 4, eg = L & 15;
      const bf16_t* src = Hb + (size_t)(k0 + n_l) * NE + e0blk + eg * 8;
      if (kt + 1 < KT) __builtin_prefetch(src + (size_t)32 * NE, 0, 1);
      u32x4 q = *(const u32x4*)src;
      const bf16_t* hh = (const bf16_t*)&q;
#pragma unroll
      for (int j = 0; j < 8; ++j) As[(eg * 8 + j) * LDT + n_l] = hh[j];
    }
#if HAVE_TDM
    if (wv == 0) __builtin_amdgcn_s_wait_tensorcnt(0);
#else
#pragma unroll
    for (int r = 0; r < 2; ++r) {
      int L = tid + r * 256, row = L >> 2, kg = L & 3;
      u32x4 q = *(const u32x4*)(Bsrc + (size_t)row * NND + k0 + kg * 8);
      *(u32x4*)(&Bs[row * LDT + kg * 8]) = q;
    }
#endif
    __syncthreads();
    v16bf af[4], bfr[2];
#pragma unroll
    for (int i = 0; i < 4; ++i) af[i] = load_frag_a(As, wm * 64 + i * 16, lane);
#pragma unroll
    for (int j = 0; j < 2; ++j) bfr[j] = load_frag_b(Bs, wn * 32 + j * 16, lane);
#pragma unroll
    for (int i = 0; i < 4; ++i)
#pragma unroll
      for (int j = 0; j < 2; ++j)
        acc[i][j] = __builtin_amdgcn_wmma_f32_16x16x32_bf16(
            false, af[i], false, bfr[j], (short)0, acc[i][j], false, false);
    __syncthreads();
  }
  int l16 = lane & 15, hi = lane >> 4;
#pragma unroll
  for (int i = 0; i < 4; ++i)
#pragma unroll
    for (int j = 0; j < 2; ++j) {
      int col = n0blk + wn * 32 + j * 16 + l16;   // hd
      int h = col >> 6;
      int e_base = e0blk + wm * 64 + i * 16 + hi * 8;
      Pack8 t;
#pragma unroll
      for (int r = 0; r < 8; ++r) {
        float s = ae[((size_t)b * NE + e_base + r) * NH + h];
        t.h[r] = (bf16_t)(acc[i][j][r] * s);
      }
      *(u32x4*)(mscT + ((size_t)b * HD + col) * NE + e_base) = t.q;
    }
}

// ---------- K5: out[b][n][hd] = Hb @ mscT^T (f32), TDM double-buffered ----------
__global__ void k_gemm_out(const bf16_t* __restrict__ Hb, const bf16_t* __restrict__ mscT,
                           float* __restrict__ out) {
  __shared__ __align__(16) bf16_t As[2 * TILE_HALVES];
  __shared__ __align__(16) bf16_t Bs[2 * TILE_HALVES];
  const int b = blockIdx.z;
  const int m0blk = blockIdx.y * 128;   // M = nodes
  const int n0blk = blockIdx.x * 128;   // N = hd
  const int tid = threadIdx.x, lane = tid & 31, wv = tid >> 5;
  const int wm = wv >> 2, wn = wv & 3;
  const v8f vz = {0.f,0.f,0.f,0.f,0.f,0.f,0.f,0.f};
  v8f acc[4][2];
#pragma unroll
  for (int i = 0; i < 4; ++i)
#pragma unroll
    for (int j = 0; j < 2; ++j) acc[i][j] = vz;
  const int KT = NE / 32;
  const bf16_t* Asrc = Hb + (size_t)m0blk * NE;
  const bf16_t* Bsrc = mscT + ((size_t)b * HD + n0blk) * NE;

#if HAVE_TDM
  if (wv == 0) {   // prologue: DMA tile 0 into buffer 0
    tdm_load_2d((unsigned)(uintptr_t)As, Asrc, NE, NND, NE, 32, 128);
    tdm_load_2d((unsigned)(uintptr_t)Bs, Bsrc, NE, HD, NE, 32, 128);
  }
  for (int kt = 0; kt < KT; ++kt) {
    const int cur = kt & 1;
    if (wv == 0) {
      if (kt + 1 < KT) {   // DMA next tile into other buffer, then wait for current
        int k1 = (kt + 1) * 32, nb = (kt + 1) & 1;
        tdm_load_2d((unsigned)(uintptr_t)(As + nb * TILE_HALVES), Asrc + k1, NE, NND, NE, 32, 128);
        tdm_load_2d((unsigned)(uintptr_t)(Bs + nb * TILE_HALVES), Bsrc + k1, NE, HD, NE, 32, 128);
        __builtin_amdgcn_s_wait_tensorcnt(2);   // in-order: current tile's 2 DMAs done
      } else {
        __builtin_amdgcn_s_wait_tensorcnt(0);
      }
    }
    __syncthreads();
    const bf16_t* Ac = As + cur * TILE_HALVES;
    const bf16_t* Bc = Bs + cur * TILE_HALVES;
    v16bf af[4], bfr[2];
#pragma unroll
    for (int i = 0; i < 4; ++i) af[i] = load_frag_a(Ac, wm * 64 + i * 16, lane);
#pragma unroll
    for (int j = 0; j < 2; ++j) bfr[j] = load_frag_b(Bc, wn * 32 + j * 16, lane);
#pragma unroll
    for (int i = 0; i < 4; ++i)
#pragma unroll
      for (int j = 0; j < 2; ++j)
        acc[i][j] = __builtin_amdgcn_wmma_f32_16x16x32_bf16(
            false, af[i], false, bfr[j], (short)0, acc[i][j], false, false);
    __syncthreads();
  }
#else
  for (int kt = 0; kt < KT; ++kt) {
    int k0 = kt * 32;
#pragma unroll
    for (int r = 0; r < 2; ++r) {
      int L = tid + r * 256, row = L >> 2, eg = L & 3;
      *(u32x4*)(&As[row * LDT + eg * 8]) =
          *(const u32x4*)(Asrc + (size_t)row * NE + k0 + eg * 8);
    }
#pragma unroll
    for (int r = 0; r < 2; ++r) {
      int L = tid + r * 256, row = L >> 2, eg = L & 3;
      *(u32x4*)(&Bs[row * LDT + eg * 8]) =
          *(const u32x4*)(Bsrc + (size_t)row * NE + k0 + eg * 8);
    }
    __syncthreads();
    v16bf af[4], bfr[2];
#pragma unroll
    for (int i = 0; i < 4; ++i) af[i] = load_frag_a(As, wm * 64 + i * 16, lane);
#pragma unroll
    for (int j = 0; j < 2; ++j) bfr[j] = load_frag_b(Bs, wn * 32 + j * 16, lane);
#pragma unroll
    for (int i = 0; i < 4; ++i)
#pragma unroll
      for (int j = 0; j < 2; ++j)
        acc[i][j] = __builtin_amdgcn_wmma_f32_16x16x32_bf16(
            false, af[i], false, bfr[j], (short)0, acc[i][j], false, false);
    __syncthreads();
  }
#endif
  int l16 = lane & 15, hi = lane >> 4;
#pragma unroll
  for (int i = 0; i < 4; ++i)
#pragma unroll
    for (int j = 0; j < 2; ++j) {
      int col = n0blk + wn * 32 + j * 16 + l16;
      int m_base = m0blk + wm * 64 + i * 16 + hi * 8;
#pragma unroll
      for (int r = 0; r < 8; ++r)
        out[((size_t)b * NND + m_base + r) * HD + col] = acc[i][j][r];
    }
}

// ---------- launch ----------
extern "C" void kernel_launch(void* const* d_in, const int* in_sizes, int n_in,
                              void* d_out, int out_size, void* d_ws, size_t ws_size,
                              hipStream_t stream) {
  const float* x = (const float*)d_in[0];
  const float* H = (const float*)d_in[1];
  const float* W = (const float*)d_in[2];
  const float* a = (const float*)d_in[3];
  float* out = (float*)d_out;

  char* ws = (char*)d_ws;
  size_t off = 0;
  auto take = [&](size_t bytes) -> char* {
    char* p = ws + off;
    off += (bytes + 255) & ~(size_t)255;
    return p;
  };
  bf16_t* Hb      = (bf16_t*)take((size_t)NND * NE * 2);        // 64 MB (L2-resident)
  bf16_t* WTb     = (bf16_t*)take((size_t)HD * FIN * 2);        // 256 KB
  bf16_t* Whb     = (bf16_t*)take((size_t)BDIM * NND * HD * 2); // 16.8 MB
  bf16_t* WhbT    = (bf16_t*)take((size_t)BDIM * HD * NND * 2); // 16.8 MB
  bf16_t* avb     = (bf16_t*)take((size_t)NND * 16 * 2);        // 256 KB
  float*  logitsT = (float*)take((size_t)16 * NE * 4);          // 256 KB
  float*  aeBuf   = (float*)take((size_t)BDIM * NE * NH * 4);   // 256 KB
  bf16_t* mscT    = (bf16_t*)take((size_t)BDIM * HD * NE * 2);  // 8.4 MB

  k_cvt_H<<<(int)(((size_t)NND * NE) / 1024), 256, 0, stream>>>(H, Hb);
  k_cvt_Wt<<<(FIN * HD) / 256, 256, 0, stream>>>(W, WTb);
  k_gemm_xW<<<dim3(HD / 128, NND / 128, BDIM), 256, 0, stream>>>(x, WTb, Whb, WhbT);
  k_av<<<(BDIM * NND * NH) / 256, 256, 0, stream>>>(Whb, a, avb);
  k_gemm_logits<<<dim3(1, NE / 128, 1), 256, 0, stream>>>(Hb, avb, logitsT);
  k_softmax<<<16, 256, 0, stream>>>(logitsT, aeBuf);
  k_gemm_m<<<dim3(HD / 128, NE / 128, BDIM), 256, 0, stream>>>(Hb, WhbT, aeBuf, mscT);
  k_gemm_out<<<dim3(HD / 128, NND / 128, BDIM), 256, 0, stream>>>(Hb, mscT, out);
}
